// MoE_85023172591908
// MI455X (gfx1250) — compile-verified, matrix operand order checked by reading, not run on hardware
//
#include <hip/hip_runtime.h>
#include <hip/hip_bf16.h>

typedef __attribute__((ext_vector_type(16))) __bf16 v16bf;
typedef __attribute__((ext_vector_type(8)))  float  v8f;
typedef int v4i __attribute__((vector_size(16)));   // matches async-builtin pointee type

#define NTOK     8192
#define DMODEL   1024
#define DFF      4096
#define NEXP     8
#define PDIM     256

#define BM 128
#define BN 128
#define BK 32
#define LSTR 40   // LDS row stride in bf16 elems (32 + 8 pad) -> 80B rows, conflict-free b128 frag reads

#if defined(__AMDGCN__) && __has_builtin(__builtin_amdgcn_global_load_async_to_lds_b128)
#define HAVE_ASYNC 1
#else
#define HAVE_ASYNC 0
#endif

#if HAVE_ASYNC
// global -> LDS async copy, 16B per lane, tracked by ASYNCcnt
typedef __attribute__((address_space(1))) v4i* gbl_v4i_p;
typedef __attribute__((address_space(3))) v4i* lds_v4i_p;
#define ASYNC_B128(gp, lp)                                                   \
  __builtin_amdgcn_global_load_async_to_lds_b128(                            \
      (gbl_v4i_p)(unsigned long long)(const void*)(gp),                      \
      (lds_v4i_p)(unsigned)(unsigned long long)(void*)(lp),                  \
      0, 0)
#if __has_builtin(__builtin_amdgcn_s_wait_asynccnt)
#define WAIT_ASYNC(n) __builtin_amdgcn_s_wait_asynccnt(n)
#else
#define WAIT_ASYNC(n) asm volatile("s_wait_asynccnt %0" ::"i"(n) : "memory")
#endif
#endif

union Frag { v16bf v; uint4 q[2]; };

// ---------------------------------------------------------------------------
// fp32 -> bf16 conversion, 4 elems/thread, grid-stride
// ---------------------------------------------------------------------------
__global__ __launch_bounds__(256) void cvt_kernel(const float4* __restrict__ src,
                                                  ushort4* __restrict__ dst, long n4) {
  long stride = (long)gridDim.x * blockDim.x;
  for (long i = (long)blockIdx.x * blockDim.x + threadIdx.x; i < n4; i += stride) {
    float4 v = src[i];
    union { __bf16 b[4]; ushort4 u; } o;
    o.b[0] = (__bf16)v.x; o.b[1] = (__bf16)v.y;
    o.b[2] = (__bf16)v.z; o.b[3] = (__bf16)v.w;
    dst[i] = o.u;
  }
}

// ---------------------------------------------------------------------------
// normalize sim_matrix columns (axis=0): [256,8] row-major
// ---------------------------------------------------------------------------
__global__ void simnorm_kernel(const float* __restrict__ sm, float* __restrict__ out) {
  int e = threadIdx.x;
  if (e < NEXP) {
    float ss = 0.f;
    for (int p = 0; p < PDIM; ++p) { float v = sm[p * NEXP + e]; ss += v * v; }
    float inv = 1.f / fmaxf(sqrtf(ss), 1e-12f);
    for (int p = 0; p < PDIM; ++p) out[p * NEXP + e] = sm[p * NEXP + e] * inv;
  }
}

// ---------------------------------------------------------------------------
// Generic bf16 WMMA GEMM:  Out = epilogue( A[rows,K] * B[N,K]^T + bias )
// MODE 0: proj   (A direct rows = tokens, store fp32)
// MODE 1: fc1    (A rows gathered via rowTok, relu, store bf16 at compact slot)
// MODE 2: fc2    (A direct compact slots from Hbf, store fp32 at compact slot)
// ---------------------------------------------------------------------------
template <int MODE>
__global__ __launch_bounds__(256)
void gemm_bf16(const __bf16* __restrict__ A, int lda,
               const __bf16* __restrict__ B, int ldb, long bStrideE,
               const float* __restrict__ bias, int biasStrideE,
               void* __restrict__ Out, int ldo, int K,
               const int* __restrict__ counts, const int* __restrict__ offsets,
               const int* __restrict__ rowTok) {
#if HAVE_ASYNC
  __shared__ alignas(16) __bf16 As[3][BM * LSTR];   // triple-buffered
  __shared__ alignas(16) __bf16 Bs[3][BN * LSTR];
#else
  __shared__ alignas(16) __bf16 As[1][BM * LSTR];
  __shared__ alignas(16) __bf16 Bs[1][BN * LSTR];
#endif

  const int e = blockIdx.z;
  int count, seg;
  if (MODE == 0) { count = NTOK; seg = 0; }
  else           { count = counts[e]; seg = offsets[e]; }

  const int row0 = blockIdx.y * BM;
  if (row0 >= count) return;
  const int n0 = blockIdx.x * BN;

  const __bf16* Bp = B + (long)e * bStrideE;
  const float*  bp = bias + (long)e * biasStrideE;

  const int tid  = threadIdx.x;
  const int lane = tid & 31;
  const int wave = tid >> 5;
  const int wrow = wave >> 2;   // 0..1 -> 64 rows each
  const int wcol = wave & 3;    // 0..3 -> 32 cols each
  const int hsel = lane >> 4;   // K-half select
  const int r16  = lane & 15;

  // tile-fill mapping: 2 threads per row, 16 bf16 (32B) each
  const int arow = tid >> 1;
  const int aseg = (tid & 1) * 16;

  const bool a_valid = (row0 + arow) < count;
  long a_src_row;
  if (MODE == 1)      a_src_row = a_valid ? (long)rowTok[seg + row0 + arow] : 0;
  else if (MODE == 2) a_src_row = a_valid ? (long)(seg + row0 + arow) : (long)seg;
  else                a_src_row = (long)(row0 + arow);

  const __bf16* aptr = A + a_src_row * lda + aseg;
  const __bf16* bptr = Bp + (long)(n0 + arow) * ldb + aseg;

  v8f acc[4][2] = {};

  auto do_wmma = [&](const __bf16* Ab, const __bf16* Bb) {
    Frag af[4], bfr[2];
#pragma unroll
    for (int mt = 0; mt < 4; ++mt) {
      const __bf16* p = Ab + (wrow * 64 + mt * 16 + r16) * LSTR + hsel * 8;
      af[mt].q[0] = *(const uint4*)(p);
      af[mt].q[1] = *(const uint4*)(p + 16);
    }
#pragma unroll
    for (int nt = 0; nt < 2; ++nt) {
      const __bf16* p = Bb + (wcol * 32 + nt * 16 + r16) * LSTR + hsel * 8;
      bfr[nt].q[0] = *(const uint4*)(p);
      bfr[nt].q[1] = *(const uint4*)(p + 16);
    }
#pragma unroll
    for (int mt = 0; mt < 4; ++mt)
#pragma unroll
      for (int nt = 0; nt < 2; ++nt)
        acc[mt][nt] = __builtin_amdgcn_wmma_f32_16x16x32_bf16(
            false, af[mt].v, false, bfr[nt].v, (short)0, acc[mt][nt], false, false);
  };

#if HAVE_ASYNC
  // async pipeline: prefetch stage s+1 while computing stage s; 1 barrier/stage.
  auto issue = [&](int k0, int buf) {
    __bf16* al = &As[buf][arow * LSTR + aseg];
    __bf16* bl = &Bs[buf][arow * LSTR + aseg];
    ASYNC_B128(aptr + k0,     al);
    ASYNC_B128(aptr + k0 + 8, al + 8);
    ASYNC_B128(bptr + k0,     bl);
    ASYNC_B128(bptr + k0 + 8, bl + 8);
  };
  const int nstage = K / BK;
  issue(0, 0);
  for (int s = 0; s < nstage; ++s) {
    if (s + 1 < nstage) {
      issue((s + 1) * BK, (s + 1) % 3);
      WAIT_ASYNC(4);   // stage s complete; stage s+1 may stay in flight
    } else {
      WAIT_ASYNC(0);
    }
    __syncthreads();   // publish stage s LDS; also WAR fence (triple buffer)
    do_wmma(As[s % 3], Bs[s % 3]);
  }
#else
  // synchronous fallback: register-staged single buffer
  for (int k0 = 0; k0 < K; k0 += BK) {
    uint4 av0, av1;
    if (a_valid) { av0 = *(const uint4*)(aptr + k0); av1 = *(const uint4*)(aptr + k0 + 8); }
    else         { av0 = make_uint4(0u,0u,0u,0u);    av1 = av0; }
    uint4 bv0 = *(const uint4*)(bptr + k0);
    uint4 bv1 = *(const uint4*)(bptr + k0 + 8);

    __syncthreads();
    *(uint4*)(As[0] + arow * LSTR + aseg)     = av0;
    *(uint4*)(As[0] + arow * LSTR + aseg + 8) = av1;
    *(uint4*)(Bs[0] + arow * LSTR + aseg)     = bv0;
    *(uint4*)(Bs[0] + arow * LSTR + aseg + 8) = bv1;
    __syncthreads();
    do_wmma(As[0], Bs[0]);
  }
#endif

  // epilogue: C layout -> elem i: m = i + 8*hsel, n = lane&15
#pragma unroll
  for (int mt = 0; mt < 4; ++mt) {
#pragma unroll
    for (int nt = 0; nt < 2; ++nt) {
      const int n = n0 + wcol * 32 + nt * 16 + r16;
      const float bv = bp[n];
#pragma unroll
      for (int i = 0; i < 8; ++i) {
        const int m = wrow * 64 + mt * 16 + hsel * 8 + i;
        const int row = row0 + m;
        if (row < count) {
          float v = acc[mt][nt][i] + bv;
          const long orow = (long)(seg + row);
          if (MODE == 1) {
            v = fmaxf(v, 0.f);
            ((__bf16*)Out)[orow * ldo + n] = (__bf16)v;
          } else {
            ((float*)Out)[orow * ldo + n] = v;
          }
        }
      }
    }
  }
}

// ---------------------------------------------------------------------------
// routing: one wave per token. l2norm(proj row) dot simnorm -> top2 + softmax
// ---------------------------------------------------------------------------
__global__ __launch_bounds__(256)
void routing_kernel(const float* __restrict__ proj, const float* __restrict__ simn,
                    const float* __restrict__ temperature,
                    int* __restrict__ eidx, float* __restrict__ gates,
                    int* __restrict__ counts) {
  __shared__ float ssim[PDIM * NEXP];
  for (int i = threadIdx.x; i < PDIM * NEXP; i += blockDim.x) ssim[i] = simn[i];
  __syncthreads();

  const int wave = threadIdx.x >> 5, lane = threadIdx.x & 31;
  const int token = blockIdx.x * 8 + wave;

  const float* pr = proj + (long)token * PDIM + lane * 8;
  float4 p0 = *(const float4*)(pr);
  float4 p1 = *(const float4*)(pr + 4);

  float ss = p0.x*p0.x + p0.y*p0.y + p0.z*p0.z + p0.w*p0.w
           + p1.x*p1.x + p1.y*p1.y + p1.z*p1.z + p1.w*p1.w;
  for (int off = 16; off; off >>= 1) ss += __shfl_xor(ss, off, 32);
  const float inv = 1.f / fmaxf(sqrtf(ss), 1e-12f);
  const float tK = expf(temperature[0]);

  float sim[NEXP];
#pragma unroll
  for (int e = 0; e < NEXP; ++e) {
    const float* s = ssim + lane * 8 * NEXP + e;
    float d = p0.x*s[0]  + p0.y*s[8]  + p0.z*s[16] + p0.w*s[24]
            + p1.x*s[32] + p1.y*s[40] + p1.z*s[48] + p1.w*s[56];
    for (int off = 16; off; off >>= 1) d += __shfl_xor(d, off, 32);
    sim[e] = d * inv * tK;
  }
  if (lane == 0) {
    int i0 = 0; float v0 = sim[0];
#pragma unroll
    for (int e = 1; e < NEXP; ++e) if (sim[e] > v0) { v0 = sim[e]; i0 = e; }
    int i1 = -1; float v1 = -3.4e38f;
#pragma unroll
    for (int e = 0; e < NEXP; ++e) if (e != i0 && sim[e] > v1) { v1 = sim[e]; i1 = e; }
    const float g0 = 1.f / (1.f + expf(v1 - v0));
    eidx[2*token]   = i0;  eidx[2*token+1]  = i1;
    gates[2*token]  = g0;  gates[2*token+1] = 1.f - g0;
    atomicAdd(&counts[i0], 1);
    atomicAdd(&counts[i1], 1);
  }
}

__global__ void init_counts(int* counts) { if (threadIdx.x < NEXP) counts[threadIdx.x] = 0; }

__global__ void scan_kernel(const int* __restrict__ counts, int* __restrict__ offsets,
                            int* __restrict__ cursors) {
  if (threadIdx.x == 0) {
    int acc = 0;
    for (int e = 0; e < NEXP; ++e) { offsets[e] = acc; cursors[e] = acc; acc += counts[e]; }
  }
}

__global__ __launch_bounds__(256)
void scatter_kernel(const int* __restrict__ eidx, int* __restrict__ cursors,
                    int* __restrict__ rowTok, int* __restrict__ tokPos) {
  const int t = blockIdx.x * blockDim.x + threadIdx.x;
  if (t < NTOK) {
    const int e0 = eidx[2*t], e1 = eidx[2*t+1];
    const int p0 = atomicAdd(&cursors[e0], 1);
    rowTok[p0] = t; tokPos[2*t] = p0;
    const int p1 = atomicAdd(&cursors[e1], 1);
    rowTok[p1] = t; tokPos[2*t+1] = p1;
  }
}

// ---------------------------------------------------------------------------
// out[t] = g0*Y[pos0] + g1*Y[pos1]   (4 floats/thread)
// ---------------------------------------------------------------------------
__global__ __launch_bounds__(256)
void combine_kernel(const float* __restrict__ Y, const float* __restrict__ gates,
                    const int* __restrict__ tokPos, float* __restrict__ out) {
  const long idx = (long)blockIdx.x * blockDim.x + threadIdx.x;  // N*D/4 threads
  const long t = idx >> 8;            // D/4 = 256
  const long c = (idx & 255) << 2;
  const float g0 = gates[2*t], g1 = gates[2*t+1];
  const long p0 = tokPos[2*t], p1 = tokPos[2*t+1];
  float4 y0 = *(const float4*)(Y + p0 * DMODEL + c);
  float4 y1 = *(const float4*)(Y + p1 * DMODEL + c);
  float4 o;
  o.x = g0*y0.x + g1*y1.x; o.y = g0*y0.y + g1*y1.y;
  o.z = g0*y0.z + g1*y1.z; o.w = g0*y0.w + g1*y1.w;
  *(float4*)(out + t * DMODEL + c) = o;
}

// ---------------------------------------------------------------------------
extern "C" void kernel_launch(void* const* d_in, const int* in_sizes, int n_in,
                              void* d_out, int out_size, void* d_ws, size_t ws_size,
                              hipStream_t stream) {
  const float* x    = (const float*)d_in[0];
  const float* cpw  = (const float*)d_in[1];
  const float* cpb  = (const float*)d_in[2];
  const float* simm = (const float*)d_in[3];
  const float* temp = (const float*)d_in[4];
  const float* f1w  = (const float*)d_in[5];
  const float* f1b  = (const float*)d_in[6];
  const float* f2w  = (const float*)d_in[7];
  const float* f2b  = (const float*)d_in[8];
  float* out = (float*)d_out;
  (void)in_sizes; (void)n_in; (void)out_size; (void)ws_size;

  char* ws = (char*)d_ws;
  size_t off = 0;
  auto alloc = [&](size_t bytes) -> char* {
    char* p = ws + off; off = (off + bytes + 255) & ~(size_t)255; return p;
  };
  __bf16* Xbf  = (__bf16*)alloc((size_t)NTOK * DMODEL * 2);
  __bf16* Wpbf = (__bf16*)alloc((size_t)PDIM * DMODEL * 2);
  __bf16* W1bf = (__bf16*)alloc((size_t)NEXP * DFF * DMODEL * 2);
  __bf16* W2bf = (__bf16*)alloc((size_t)NEXP * DMODEL * DFF * 2);
  float*  proj = (float*)alloc((size_t)NTOK * PDIM * 4);
  float*  simn = (float*)alloc((size_t)PDIM * NEXP * 4);
  int*    ctl  = (int*)alloc(1024);
  int*    eidx = (int*)alloc((size_t)NTOK * 2 * 4);
  float*  gts  = (float*)alloc((size_t)NTOK * 2 * 4);
  int*    tokP = (int*)alloc((size_t)NTOK * 2 * 4);
  int*    rowT = (int*)alloc((size_t)NTOK * 2 * 4);
  __bf16* Hbf  = (__bf16*)alloc((size_t)2 * NTOK * DFF * 2);
  float*  Yws  = (float*)alloc((size_t)2 * NTOK * DMODEL * 4);

  int* counts = ctl; int* offsets = ctl + 8; int* cursors = ctl + 16;

  // 1) bf16 conversions
  cvt_kernel<<<4096, 256, 0, stream>>>((const float4*)x,   (ushort4*)Xbf,  (long)NTOK * DMODEL / 4);
  cvt_kernel<<<256,  256, 0, stream>>>((const float4*)cpw, (ushort4*)Wpbf, (long)PDIM * DMODEL / 4);
  cvt_kernel<<<8192, 256, 0, stream>>>((const float4*)f1w, (ushort4*)W1bf, (long)NEXP * DFF * DMODEL / 4);
  cvt_kernel<<<8192, 256, 0, stream>>>((const float4*)f2w, (ushort4*)W2bf, (long)NEXP * DMODEL * DFF / 4);
  simnorm_kernel<<<1, 64, 0, stream>>>(simm, simn);

  // 2) gate projection GEMM (WMMA)
  gemm_bf16<0><<<dim3(PDIM / BN, NTOK / BM, 1), 256, 0, stream>>>(
      Xbf, DMODEL, Wpbf, DMODEL, 0, cpb, 0, proj, PDIM, DMODEL,
      nullptr, nullptr, nullptr);

  // 3) routing + list build
  init_counts<<<1, 32, 0, stream>>>(counts);
  routing_kernel<<<NTOK / 8, 256, 0, stream>>>(proj, simn, temp, eidx, gts, counts);
  scan_kernel<<<1, 32, 0, stream>>>(counts, offsets, cursors);
  scatter_kernel<<<NTOK / 256, 256, 0, stream>>>(eidx, cursors, rowT, tokP);

  // 4) fc1 (gathered rows, relu, bf16 H)
  gemm_bf16<1><<<dim3(DFF / BN, NTOK / BM, NEXP), 256, 0, stream>>>(
      Xbf, DMODEL, W1bf, DMODEL, (long)DFF * DMODEL, f1b, DFF,
      Hbf, DFF, DMODEL, counts, offsets, rowT);

  // 5) fc2 (compact rows, fp32 Y)
  gemm_bf16<2><<<dim3(DMODEL / BN, NTOK / BM, NEXP), 256, 0, stream>>>(
      Hbf, DFF, W2bf, DFF, (long)DMODEL * DFF, f2b, DMODEL,
      Yws, DMODEL, DFF, counts, offsets, rowT);

  // 6) gate-weighted combine
  combine_kernel<<<(NTOK * DMODEL / 4) / 256, 256, 0, stream>>>(Yws, gts, tokP, out);
}
